// StegoDiSModel_2061584302199
// MI455X (gfx1250) — matrix operand order repaired and checked by reading, not compiled, not run on hardware
//
#include <hip/hip_runtime.h>

// ---------------------------------------------------------------------------
// StegoDiS forward for MI455X (gfx1250, wave32, WMMA).
// GEMMs: f16 A/B fragments, f32 accumulate via v_wmma_f32_16x16x32_f16,
//        2x2 register blocking per wave (32x32 output) for 2x L2-traffic cut.
// Scan: 16 lanes per (b,d) channel, shfl_xor reduction over states.
// ---------------------------------------------------------------------------

typedef _Float16 h16;
typedef __attribute__((ext_vector_type(16))) _Float16 v16h;
typedef __attribute__((ext_vector_type(8)))  _Float16 v8h;
typedef __attribute__((ext_vector_type(8)))  float    v8f;

union AB16 { v16h v; v8h h[2]; };

// Model constants
#define BSZ 8
#define LSEQ 256
#define BL 2048      // BSZ*LSEQ
#define DM 192
#define DI 384
#define DX 768       // 2*DI
#define DRN 48       // padded DR(12)+N(16)+N(16)=44 -> 48
#define DEPTH 12

__device__ __forceinline__ float siluf(float x)   { return x / (1.0f + __expf(-x)); }
__device__ __forceinline__ float sigmf(float x)   { return 1.0f / (1.0f + __expf(-x)); }

// ---------------------------------------------------------------------------
// Epilogue helper: stores one 16x16 f32 tile held in a v8f accumulator.
//  0: store   1: +bias   2: +bias+pos[l]   3: sigmoid(+bias)
//  4: +bias+alpha*p1[b]   5: C += p1[b]*v (gate add)
//  6: C += p1[b]*v*silu(p2[row])*p3[row]
// ---------------------------------------------------------------------------
__device__ __forceinline__ void sdm_epi_store(
    float* __restrict__ C, v8f acc, int m0, int n, int lane, int ldc,
    const float* __restrict__ bias,
    const float* __restrict__ p1, int p1s,
    const float* __restrict__ p2, const float* __restrict__ p3,
    float alpha, int epi)
{
    const float bv = bias ? bias[n] : 0.0f;
#pragma unroll
    for (int r = 0; r < 8; ++r) {
        const int m = m0 + r + ((lane >> 4) << 3);
        const int b = m >> 8;                    // row / LSEQ
        float v = acc[r];
        float* cp = C + (size_t)m * ldc + n;
        switch (epi) {
            case 0: *cp = v; break;
            case 1: *cp = v + bv; break;
            case 2: *cp = v + bv + p1[(m & 255) * DM + n]; break;
            case 3: *cp = sigmf(v + bv); break;
            case 4: *cp = v + bv + alpha * p1[b * p1s + n]; break;
            case 5: *cp += p1[b * p1s + n] * v; break;
            case 6: {
                float z = p2[(size_t)m * DM + n];
                *cp += p1[b * p1s + n] * v * siluf(z) * p3[(size_t)m * DM + n];
            } break;
        }
    }
}

// ---------------------------------------------------------------------------
// WMMA GEMM:  C[M,N] = A[M,K](f16) * Bw[N,K]^T(f16) + epilogue.
// One wave computes a 32x32 block (2x2 tiles of 16x16): per K-step it loads
// 2 A + 2 B fragments and issues 4 WMMAs -> half the L2 traffic per WMMA of
// the naive 1-tile scheme.  M must be a multiple of 32; ragged N guarded.
// ---------------------------------------------------------------------------
__global__ __launch_bounds__(256) void sdm_gemm_wmma(
    const h16* __restrict__ A, const h16* __restrict__ Bw, float* __restrict__ C,
    int M, int N, int K, int lda, int ldb, int ldc,
    const float* __restrict__ bias,
    const float* __restrict__ p1, int p1s,
    const float* __restrict__ p2, const float* __restrict__ p3,
    const float* __restrict__ alphap, int aidx, int epi)
{
    const int lane    = threadIdx.x & 31;
    const int wave    = threadIdx.x >> 5;
    const int tn_cnt2 = (N + 31) >> 5;           // N 32-blocks (may be ragged)
    const int tiles   = (M >> 5) * tn_cnt2;      // M 32-blocks x N 32-blocks
    const int tid     = blockIdx.x * 8 + wave;
    if (tid >= tiles) return;
    const int tm = tid / tn_cnt2, tn = tid - tm * tn_cnt2;
    const int m0 = tm << 5, n0 = tn << 5;
    const bool n1ok = (n0 + 16) < N;             // second N tile valid?

    // A fragment: lane -> row m+(lane&15); half elems 0..7 = K base, 8..15 = +16
    const int ak = (lane >> 4) << 3;             // 0 or 8
    // B fragment: lane -> col n+(lane&15); half elems 0..7 = K base, 8..15 = +16
    const int bk = (lane >> 4) << 3;             // 0 or 8

    const h16* ap0 = A  + (size_t)(m0 + (lane & 15)) * lda + ak;
    const h16* ap1 = ap0 + (size_t)16 * lda;
    const h16* bp0 = Bw + (size_t)(n0 + (lane & 15)) * ldb + bk;
    const h16* bp1 = n1ok ? bp0 + (size_t)16 * ldb : bp0;

    v8f acc00 = {}, acc01 = {}, acc10 = {}, acc11 = {};
    for (int k0 = 0; k0 < K; k0 += 32) {
        AB16 a0, a1, b0, b1;
        a0.h[0] = *(const v8h*)(ap0);  a0.h[1] = *(const v8h*)(ap0 + 16);
        a1.h[0] = *(const v8h*)(ap1);  a1.h[1] = *(const v8h*)(ap1 + 16);
        b0.h[0] = *(const v8h*)(bp0);  b0.h[1] = *(const v8h*)(bp0 + 16);
        b1.h[0] = *(const v8h*)(bp1);  b1.h[1] = *(const v8h*)(bp1 + 16);
        ap0 += 32; ap1 += 32; bp0 += 32; bp1 += 32;
        if (k0 + 32 < K) {
            __builtin_prefetch(ap0, 0, 3); __builtin_prefetch(ap1, 0, 3);
            __builtin_prefetch(bp0, 0, 3); __builtin_prefetch(bp1, 0, 3);
        }
        acc00 = __builtin_amdgcn_wmma_f32_16x16x32_f16(false, a0.v, false, b0.v,
                                                       (short)0, acc00, false, false);
        acc01 = __builtin_amdgcn_wmma_f32_16x16x32_f16(false, a0.v, false, b1.v,
                                                       (short)0, acc01, false, false);
        acc10 = __builtin_amdgcn_wmma_f32_16x16x32_f16(false, a1.v, false, b0.v,
                                                       (short)0, acc10, false, false);
        acc11 = __builtin_amdgcn_wmma_f32_16x16x32_f16(false, a1.v, false, b1.v,
                                                       (short)0, acc11, false, false);
    }

    const float alpha = alphap ? alphap[aidx] : 0.0f;
    const int   nlo   = n0 + (lane & 15);
    sdm_epi_store(C, acc00, m0,      nlo,      lane, ldc, bias, p1, p1s, p2, p3, alpha, epi);
    sdm_epi_store(C, acc10, m0 + 16, nlo,      lane, ldc, bias, p1, p1s, p2, p3, alpha, epi);
    if (n1ok) {
        sdm_epi_store(C, acc01, m0,      nlo + 16, lane, ldc, bias, p1, p1s, p2, p3, alpha, epi);
        sdm_epi_store(C, acc11, m0 + 16, nlo + 16, lane, ldc, bias, p1, p1s, p2, p3, alpha, epi);
    }
}

// ---------------------------------------------------------------------------
// f32 -> f16 convert
// ---------------------------------------------------------------------------
__global__ __launch_bounds__(256) void sdm_f32_to_f16(const float* __restrict__ s,
                                                      h16* __restrict__ d, int n)
{
    int i = blockIdx.x * 256 + threadIdx.x;
    if (i < n) d[i] = (h16)s[i];
}

// Pad x_proj weights [12,44,384] -> f16 [12,48,384] (zero rows 44..47)
__global__ __launch_bounds__(256) void sdm_pack_xproj(const float* __restrict__ s,
                                                      h16* __restrict__ d)
{
    int i = blockIdx.x * 256 + threadIdx.x;
    if (i >= DEPTH * DRN * DI) return;
    int l = i / (DRN * DI); int r = (i / DI) % DRN; int k = i % DI;
    d[i] = (h16)((r < 44) ? s[((size_t)l * 44 + r) * DI + k] : 0.0f);
}

// ---------------------------------------------------------------------------
// Haar DWT + patchify: builds f16 patch matrices for sem [2048,192] and
// tex [2048,576] directly from x [8,3,256,256].
// ---------------------------------------------------------------------------
__global__ __launch_bounds__(256) void sdm_patchify(const float* __restrict__ x,
                                                    h16* __restrict__ psem,
                                                    h16* __restrict__ ptex)
{
    const int TS = BL * DM, TT = BL * 576;
    int i = blockIdx.x * 256 + threadIdx.x;
    if (i >= TS + TT) return;
    int row, f, c, band, p;
    bool sem = (i < TS);
    if (sem) { row = i / DM;  f = i % DM;  c = f >> 6; band = 0;          p = f & 63; }
    else     { int j = i - TS; row = j / 576; f = j % 576;
               int ch9 = f >> 6; c = ch9 % 3; band = ch9 / 3 + 1;         p = f & 63; }
    int b = row >> 8, l = row & 255;
    int ph = l >> 4, pw = l & 15;
    int py = p >> 3, px = p & 7;
    int ii = ph * 8 + py, jj = pw * 8 + px;
    const float* xb = x + ((size_t)(b * 3 + c)) * 65536;
    float a  = xb[(2*ii)  *256 + 2*jj    ];
    float b2 = xb[(2*ii)  *256 + 2*jj + 1];
    float c2 = xb[(2*ii+1)*256 + 2*jj    ];
    float d2 = xb[(2*ii+1)*256 + 2*jj + 1];
    float v;
    switch (band) {
        case 0:  v = a + b2 + c2 + d2; break;  // ll
        case 1:  v = a - b2 + c2 - d2; break;  // lh
        case 2:  v = a + b2 - c2 - d2; break;  // hl
        default: v = a - b2 - c2 + d2; break;  // hh
    }
    v *= 0.5f;
    if (sem) psem[i] = (h16)v; else ptex[i - TS] = (h16)v;
}

// ---------------------------------------------------------------------------
// Timestep embedding + c MLP (tiny; single block, staged through LDS)
// ---------------------------------------------------------------------------
__global__ __launch_bounds__(256) void sdm_timec(
    const float* __restrict__ t,
    const float* __restrict__ tw1, const float* __restrict__ tb1,
    const float* __restrict__ tw2, const float* __restrict__ tb2,
    float* __restrict__ cbuf, float* __restrict__ scbuf)
{
    __shared__ float temb[BSZ * DM];
    __shared__ float hid[BSZ * DX];
    int tid = threadIdx.x;
    for (int i = tid; i < BSZ * DM; i += 256) {
        int b = i / DM, j = i % DM;
        int k = (j < 96) ? j : j - 96;
        float fr  = __expf(-logf(10000.0f) * (float)k / 96.0f);
        float arg = t[b] * fr;
        temb[i] = (j < 96) ? cosf(arg) : sinf(arg);
    }
    __syncthreads();
    for (int i = tid; i < BSZ * DX; i += 256) {
        int b = i / DX, o = i % DX;
        float acc = tb1[o];
        for (int j = 0; j < DM; ++j) acc += temb[b * DM + j] * tw1[o * DM + j];
        hid[i] = siluf(acc);
    }
    __syncthreads();
    for (int i = tid; i < BSZ * DM; i += 256) {
        int b = i / DM, o = i % DM;
        float acc = tb2[o];
        for (int j = 0; j < DX; ++j) acc += hid[b * DX + j] * tw2[o * DX + j];
        cbuf[i]  = acc;
        scbuf[i] = siluf(acc);
    }
}

// ---------------------------------------------------------------------------
// Precompute all conditioning vectors: adaLN mods per layer, secret tanh
// projections per layer, final-layer mods. (tiny GEMVs, VALU)
// ---------------------------------------------------------------------------
__global__ __launch_bounds__(256) void sdm_mods(
    const float* __restrict__ sc, const float* __restrict__ secret,
    const float* __restrict__ adaw_s, const float* __restrict__ adab_s,
    const float* __restrict__ adaw_t, const float* __restrict__ adab_t,
    const float* __restrict__ sxw, const float* __restrict__ sxb,
    const float* __restrict__ szw, const float* __restrict__ szb,
    const float* __restrict__ fsaw, const float* __restrict__ fsab,
    const float* __restrict__ ftaw, const float* __restrict__ ftab,
    float* __restrict__ msem, float* __restrict__ mtex,
    float* __restrict__ sxt,  float* __restrict__ szt,
    float* __restrict__ fsm,  float* __restrict__ ftm)
{
    const int NA = DEPTH * BSZ * 576, NS = DEPTH * BSZ * DM, NF = BSZ * 384;
    int i = blockIdx.x * 256 + threadIdx.x;
    if (i < NA) {
        int l = i / (BSZ*576); int r = i % (BSZ*576); int b = r / 576; int o = r % 576;
        const float* w = adaw_s + ((size_t)l*576 + o) * DM;
        float acc = adab_s[l*576 + o];
        for (int k = 0; k < DM; ++k) acc += sc[b*DM + k] * w[k];
        msem[i] = acc; return;
    }
    i -= NA;
    if (i < NA) {
        int l = i / (BSZ*576); int r = i % (BSZ*576); int b = r / 576; int o = r % 576;
        const float* w = adaw_t + ((size_t)l*576 + o) * DM;
        float acc = adab_t[l*576 + o];
        for (int k = 0; k < DM; ++k) acc += sc[b*DM + k] * w[k];
        mtex[i] = acc; return;
    }
    i -= NA;
    if (i < NS) {
        int l = i / (BSZ*DM); int r = i % (BSZ*DM); int b = r / DM; int o = r % DM;
        const float* w = sxw + ((size_t)l*DM + o) * 512;
        float acc = sxb[l*DM + o];
        for (int k = 0; k < 512; ++k) acc += secret[b*512 + k] * w[k];
        sxt[i] = tanhf(acc); return;
    }
    i -= NS;
    if (i < NS) {
        int l = i / (BSZ*DM); int r = i % (BSZ*DM); int b = r / DM; int o = r % DM;
        const float* w = szw + ((size_t)l*DM + o) * 512;
        float acc = szb[l*DM + o];
        for (int k = 0; k < 512; ++k) acc += secret[b*512 + k] * w[k];
        szt[i] = tanhf(acc); return;
    }
    i -= NS;
    if (i < NF) {
        int b = i / 384, o = i % 384;
        float acc = fsab[o];
        for (int k = 0; k < DM; ++k) acc += sc[b*DM + k] * fsaw[o*DM + k];
        fsm[i] = acc; return;
    }
    i -= NF;
    if (i < NF) {
        int b = i / 384, o = i % 384;
        float acc = ftab[o];
        for (int k = 0; k < DM; ++k) acc += sc[b*DM + k] * ftaw[o*DM + k];
        ftm[i] = acc;
    }
}

// ---------------------------------------------------------------------------
// RMSNorm + adaLN modulate (one wave per row of [2048,192]).
// outA = f16(mod);  outB (optional) = f16(mod + alpha*addv[b])   (tex xt path)
// ---------------------------------------------------------------------------
__global__ __launch_bounds__(256) void sdm_modulate_rms(
    const float* __restrict__ h, const float* __restrict__ normw,
    const float* __restrict__ mod, h16* __restrict__ outA, h16* __restrict__ outB,
    const float* __restrict__ addv, const float* __restrict__ alphap, int aidx)
{
    int row  = blockIdx.x * 8 + (threadIdx.x >> 5);
    int lane = threadIdx.x & 31;
    if (row >= BL) return;
    const float* hr = h + (size_t)row * DM;
    float vals[6]; float ss = 0.0f;
#pragma unroll
    for (int i = 0; i < 6; ++i) { float v = hr[lane + 32*i]; vals[i] = v; ss += v*v; }
#pragma unroll
    for (int o = 16; o > 0; o >>= 1) ss += __shfl_xor(ss, o, 32);
    float r = rsqrtf(ss * (1.0f/192.0f) + 1e-5f);
    int b = row >> 8;
    float alpha = alphap ? alphap[aidx] : 0.0f;
#pragma unroll
    for (int i = 0; i < 6; ++i) {
        int j = lane + 32*i;
        float m = vals[i]*r*normw[j]*(1.0f + mod[b*576 + DM + j]) + mod[b*576 + j];
        outA[(size_t)row*DM + j] = (h16)m;
        if (outB) outB[(size_t)row*DM + j] = (h16)(m + alpha * addv[b*DM + j]);
    }
}

// LayerNorm + final adaLN modulate, mod layout [8,384] (sh | sc)
__global__ __launch_bounds__(256) void sdm_final_ln(
    const float* __restrict__ h, const float* __restrict__ mod, h16* __restrict__ out16)
{
    int row  = blockIdx.x * 8 + (threadIdx.x >> 5);
    int lane = threadIdx.x & 31;
    if (row >= BL) return;
    const float* hr = h + (size_t)row * DM;
    float vals[6]; float s = 0.0f;
#pragma unroll
    for (int i = 0; i < 6; ++i) { vals[i] = hr[lane + 32*i]; s += vals[i]; }
#pragma unroll
    for (int o = 16; o > 0; o >>= 1) s += __shfl_xor(s, o, 32);
    float mu = s * (1.0f/192.0f);
    float vv = 0.0f;
#pragma unroll
    for (int i = 0; i < 6; ++i) { float d = vals[i] - mu; vv += d*d; }
#pragma unroll
    for (int o = 16; o > 0; o >>= 1) vv += __shfl_xor(vv, o, 32);
    float rin = rsqrtf(vv * (1.0f/192.0f) + 1e-6f);
    int b = row >> 8;
#pragma unroll
    for (int i = 0; i < 6; ++i) {
        int j = lane + 32*i;
        float xn = (vals[i]-mu)*rin*(1.0f + mod[b*384 + DM + j]) + mod[b*384 + j];
        out16[(size_t)row*DM + j] = (h16)xn;
    }
}

// ---------------------------------------------------------------------------
// Causal conv1d (K=4) + SiLU; dir=1 applies reversed-sequence conv in place
// (anti-causal taps), output stored at original indices.
// ---------------------------------------------------------------------------
__global__ __launch_bounds__(256) void sdm_conv_silu(
    const float* __restrict__ xz, const float* __restrict__ w,
    const float* __restrict__ bconv, float* __restrict__ xc,
    h16* __restrict__ xc16, int dir)
{
    int i = blockIdx.x * 256 + threadIdx.x;
    if (i >= BL * DI) return;
    int row = i / DI, d = i % DI;
    int b = row >> 8, t = row & 255;
    float acc = bconv[d];
#pragma unroll
    for (int k = 0; k < 4; ++k) {
        int tt = dir ? (t + 3 - k) : (t - 3 + k);
        if (tt >= 0 && tt < LSEQ) acc += w[d*4 + k] * xz[((size_t)(b*LSEQ + tt))*DX + d];
    }
    float v = siluf(acc);
    xc[i] = v; xc16[i] = (h16)v;
}

// dt = softplus(dt_in @ dt_w^T + dt_b)   (K=12, VALU)
__global__ __launch_bounds__(256) void sdm_dtproj(
    const float* __restrict__ xdbl, const float* __restrict__ dtw,
    const float* __restrict__ dtb, float* __restrict__ dt)
{
    int i = blockIdx.x * 256 + threadIdx.x;
    if (i >= BL * DI) return;
    int row = i / DI, d = i % DI;
    float acc = dtb[d];
#pragma unroll
    for (int r = 0; r < 12; ++r) acc += xdbl[(size_t)row*DRN + r] * dtw[d*12 + r];
    dt[i] = (acc > 20.0f) ? acc : log1pf(__expf(acc));
}

// ---------------------------------------------------------------------------
// Selective scan: each 16-lane group owns one (b,d) channel, lane n owns
// state n.  y reduction via shfl_xor within the 16-lane group (wave32).
// Output gated by silu(z) and accumulated into ytot (dir1 adds).
// ---------------------------------------------------------------------------
__global__ __launch_bounds__(256) void sdm_scan(
    const float* __restrict__ xc, const float* __restrict__ dt,
    const float* __restrict__ xdbl, const float* __restrict__ Alog,
    const float* __restrict__ Dp, const float* __restrict__ xz,
    float* __restrict__ ytot, int dir, int accum)
{
    int half = (blockIdx.x * 256 + threadIdx.x) >> 4;   // channel index
    int n = threadIdx.x & 15;
    if (half >= BSZ * DI) return;
    int b = half / DI, d = half % DI;
    float A_ = -__expf(Alog[d*16 + n]);
    float D_ = Dp[d];
    float h = 0.0f;
    int base = b * LSEQ;
    for (int s = 0; s < LSEQ; ++s) {
        int t = dir ? (LSEQ - 1 - s) : s;
        size_t row = (size_t)(base + t);
        float dtv = dt[row*DI + d];
        float xv  = xc[row*DI + d];
        float Bv  = xdbl[row*DRN + 12 + n];
        float Cv  = xdbl[row*DRN + 28 + n];
        float dA  = __expf(dtv * A_);
        h = dA * h + (dtv * xv) * Bv;
        float p = h * Cv;
        p += __shfl_xor(p, 1, 16);
        p += __shfl_xor(p, 2, 16);
        p += __shfl_xor(p, 4, 16);
        p += __shfl_xor(p, 8, 16);
        if (n == 0) {
            float zv = xz[row*DX + DI + d];
            float y  = (p + xv * D_) * siluf(zv);
            if (accum) ytot[row*DI + d] += y; else ytot[row*DI + d] = y;
        }
    }
}

// ---------------------------------------------------------------------------
// Unpatchify + inverse Haar DWT -> d_out [8,3,256,256]
// ---------------------------------------------------------------------------
__global__ __launch_bounds__(256) void sdm_unpatch_idwt(
    const float* __restrict__ osem, const float* __restrict__ otex,
    float* __restrict__ out)
{
    int i = blockIdx.x * 256 + threadIdx.x;
    if (i >= BSZ * 3 * 256 * 256) return;
    int w = i & 255; int hh = (i >> 8) & 255; int c = (i >> 16) % 3; int b = i / (3 * 65536);
    int ii = hh >> 1, jj = w >> 1, dy = hh & 1, dx = w & 1;
    int p1 = ii & 7, hp1 = ii >> 3, p2 = jj & 7, hp2 = jj >> 3;
    size_t row  = (size_t)(b * LSEQ + hp1 * 16 + hp2);
    int    pidx = p1 * 8 + p2;
    float ll = osem[row*DM  + pidx*3 + c];
    float lh = otex[row*576 + pidx*9 + 0 + c];
    float hl = otex[row*576 + pidx*9 + 3 + c];
    float hq = otex[row*576 + pidx*9 + 6 + c];
    float v;
    if      (!dy && !dx) v = ll + lh + hl + hq;
    else if (!dy &&  dx) v = ll - lh + hl - hq;
    else if ( dy && !dx) v = ll + lh - hl - hq;
    else                 v = ll - lh - hl + hq;
    out[i] = 0.5f * v;
}

// ---------------------------------------------------------------------------
// Host-side orchestration
// ---------------------------------------------------------------------------
enum {
    IN_X = 0, IN_T, IN_SECRET,
    IN_SEM_PW, IN_SEM_PB, IN_TEX_PW, IN_TEX_PB, IN_POS,
    IN_TW1, IN_TB1, IN_TW2, IN_TB2,
    IN_FS_ADA_W, IN_FS_ADA_B, IN_FS_LIN_W, IN_FS_LIN_B,
    IN_FT_ADA_W, IN_FT_ADA_B, IN_FT_LIN_W, IN_FT_LIN_B,
    IN_NORM_SEM_W, IN_NORM_TEX_W,
    IN_ADA_SEM_W, IN_ADA_SEM_B, IN_ADA_TEX_W, IN_ADA_TEX_B,
    IN_MASK_W, IN_MASK_B, IN_TEXZ_W, IN_TEXZ_B,
    IN_SX_W, IN_SX_B, IN_SZ_W, IN_SZ_B,
    IN_ALPHA_X, IN_ALPHA_Z,
    IN_SEM_IN_W, IN_SEM_OUT_W,
    IN_SEM_CONV_W_F, IN_SEM_CONV_B_F, IN_SEM_XPROJ_W_F,
    IN_SEM_DT_W_F, IN_SEM_DT_B_F, IN_SEM_ALOG_F, IN_SEM_D_F,
    IN_SEM_CONV_W_B, IN_SEM_CONV_B_B, IN_SEM_XPROJ_W_B,
    IN_SEM_DT_W_B, IN_SEM_DT_B_B, IN_SEM_ALOG_B, IN_SEM_D_B,
    IN_TEX_IN_W, IN_TEX_OUT_W,
    IN_TEX_CONV_W_F, IN_TEX_CONV_B_F, IN_TEX_XPROJ_W_F,
    IN_TEX_DT_W_F, IN_TEX_DT_B_F, IN_TEX_ALOG_F, IN_TEX_D_F,
    IN_TEX_CONV_W_B, IN_TEX_CONV_B_B, IN_TEX_XPROJ_W_B,
    IN_TEX_DT_W_B, IN_TEX_DT_B_B, IN_TEX_ALOG_B, IN_TEX_D_B
};

extern "C" void kernel_launch(void* const* d_in, const int* in_sizes, int n_in,
                              void* d_out, int out_size, void* d_ws, size_t ws_size,
                              hipStream_t stream)
{
    (void)in_sizes; (void)n_in; (void)out_size; (void)ws_size;
    char* base = (char*)d_ws;
    size_t off = 0;
    auto alloc = [&](size_t elems, size_t esz) -> void* {
        void* p = base + off;
        off += (elems * esz + 255) & ~(size_t)255;
        return p;
    };
    // f16 weights
    h16* Wsem_p = (h16*)alloc(DM*DM, 2);
    h16* Wtex_p = (h16*)alloc(DM*576, 2);
    h16* Win_s  = (h16*)alloc((size_t)DEPTH*DX*DM, 2);
    h16* Win_t  = (h16*)alloc((size_t)DEPTH*DX*DM, 2);
    h16* Wout_s = (h16*)alloc((size_t)DEPTH*DM*DI, 2);
    h16* Wout_t = (h16*)alloc((size_t)DEPTH*DM*DI, 2);
    h16* Wx     = (h16*)alloc((size_t)4*DEPTH*DRN*DI, 2);
    h16* Wmask  = (h16*)alloc((size_t)DEPTH*DM*DM, 2);
    h16* Wtexz  = (h16*)alloc((size_t)DEPTH*DM*DM, 2);
    h16* Wfs    = (h16*)alloc(DM*DM, 2);
    h16* Wft    = (h16*)alloc(576*DM, 2);
    // f16 activations
    h16* psem16 = (h16*)alloc((size_t)BL*DM, 2);
    h16* ptex16 = (h16*)alloc((size_t)BL*576, 2);
    h16* u16    = (h16*)alloc((size_t)BL*DM, 2);
    h16* xt16   = (h16*)alloc((size_t)BL*DM, 2);
    h16* tm16   = (h16*)alloc((size_t)BL*DM, 2);
    h16* hs16   = (h16*)alloc((size_t)BL*DM, 2);
    h16* fin16  = (h16*)alloc((size_t)BL*DM, 2);
    h16* xc16   = (h16*)alloc((size_t)BL*DI, 2);
    h16* y16    = (h16*)alloc((size_t)BL*DI, 2);
    // f32 buffers
    float* h_sem = (float*)alloc((size_t)BL*DM, 4);
    float* h_tex = (float*)alloc((size_t)BL*DM, 4);
    float* xz    = (float*)alloc((size_t)BL*DX, 4);
    float* xc    = (float*)alloc((size_t)BL*DI, 4);
    float* dtf   = (float*)alloc((size_t)BL*DI, 4);
    float* ytot  = (float*)alloc((size_t)BL*DI, 4);
    float* xdbl  = (float*)alloc((size_t)BL*DRN, 4);
    float* ztb   = (float*)alloc((size_t)BL*DM, 4);
    float* maskb = (float*)alloc((size_t)BL*DM, 4);
    float* cbuf  = (float*)alloc(BSZ*DM, 4);
    float* scbuf = (float*)alloc(BSZ*DM, 4);
    float* msem  = (float*)alloc((size_t)DEPTH*BSZ*576, 4);
    float* mtex  = (float*)alloc((size_t)DEPTH*BSZ*576, 4);
    float* sxt   = (float*)alloc((size_t)DEPTH*BSZ*DM, 4);
    float* szt   = (float*)alloc((size_t)DEPTH*BSZ*DM, 4);
    float* fsm   = (float*)alloc(BSZ*384, 4);
    float* ftm   = (float*)alloc(BSZ*384, 4);
    float* osem  = (float*)alloc((size_t)BL*DM, 4);
    float* otex  = (float*)alloc((size_t)BL*576, 4);

    auto F = [&](int idx) { return (const float*)d_in[idx]; };
    auto cvt = [&](int idx, h16* dst, int n) {
        sdm_f32_to_f16<<<(n + 255) / 256, 256, 0, stream>>>(F(idx), dst, n);
    };
    auto gemm = [&](const h16* A, const h16* Bw, float* C, int M, int N, int K,
                    int lda, int ldb, int ldc, const float* bias,
                    const float* p1, int p1s, const float* p2, const float* p3,
                    const float* alphap, int aidx, int epi) {
        int tiles = (M / 32) * ((N + 31) / 32);   // 32x32 blocks per wave
        sdm_gemm_wmma<<<(tiles + 7) / 8, 256, 0, stream>>>(
            A, Bw, C, M, N, K, lda, ldb, ldc, bias, p1, p1s, p2, p3, alphap, aidx, epi);
    };

    // ---- weight prep (f32 -> f16, deterministic every call) ----
    cvt(IN_SEM_PW, Wsem_p, DM*DM);
    cvt(IN_TEX_PW, Wtex_p, DM*576);
    cvt(IN_SEM_IN_W,  Win_s,  DEPTH*DX*DM);
    cvt(IN_TEX_IN_W,  Win_t,  DEPTH*DX*DM);
    cvt(IN_SEM_OUT_W, Wout_s, DEPTH*DM*DI);
    cvt(IN_TEX_OUT_W, Wout_t, DEPTH*DM*DI);
    cvt(IN_MASK_W, Wmask, DEPTH*DM*DM);
    cvt(IN_TEXZ_W, Wtexz, DEPTH*DM*DM);
    cvt(IN_FS_LIN_W, Wfs, DM*DM);
    cvt(IN_FT_LIN_W, Wft, 576*DM);
    {
        int nblk = (DEPTH*DRN*DI + 255) / 256;
        sdm_pack_xproj<<<nblk, 256, 0, stream>>>(F(IN_SEM_XPROJ_W_F), Wx + (size_t)0*DEPTH*DRN*DI);
        sdm_pack_xproj<<<nblk, 256, 0, stream>>>(F(IN_SEM_XPROJ_W_B), Wx + (size_t)1*DEPTH*DRN*DI);
        sdm_pack_xproj<<<nblk, 256, 0, stream>>>(F(IN_TEX_XPROJ_W_F), Wx + (size_t)2*DEPTH*DRN*DI);
        sdm_pack_xproj<<<nblk, 256, 0, stream>>>(F(IN_TEX_XPROJ_W_B), Wx + (size_t)3*DEPTH*DRN*DI);
    }

    // ---- DWT + patchify + patch embed ----
    sdm_patchify<<<(BL*DM + BL*576 + 255) / 256, 256, 0, stream>>>(F(IN_X), psem16, ptex16);
    gemm(psem16, Wsem_p, h_sem, BL, DM, DM, DM, DM, DM,
         F(IN_SEM_PB), F(IN_POS), 0, nullptr, nullptr, nullptr, 0, 2);
    gemm(ptex16, Wtex_p, h_tex, BL, DM, 576, 576, 576, DM,
         F(IN_TEX_PB), F(IN_POS), 0, nullptr, nullptr, nullptr, 0, 2);

    // ---- conditioning ----
    sdm_timec<<<1, 256, 0, stream>>>(F(IN_T), F(IN_TW1), F(IN_TB1), F(IN_TW2), F(IN_TB2),
                                     cbuf, scbuf);
    sdm_mods<<<600, 256, 0, stream>>>(scbuf, F(IN_SECRET),
        F(IN_ADA_SEM_W), F(IN_ADA_SEM_B), F(IN_ADA_TEX_W), F(IN_ADA_TEX_B),
        F(IN_SX_W), F(IN_SX_B), F(IN_SZ_W), F(IN_SZ_B),
        F(IN_FS_ADA_W), F(IN_FS_ADA_B), F(IN_FT_ADA_W), F(IN_FT_ADA_B),
        msem, mtex, sxt, szt, fsm, ftm);

    // ---- mamba branch runner ----
    auto run_mamba = [&](const h16* Ain, const h16* Win, const h16* Wout,
                         int convw_i, int convb_i, int xslot,
                         int dtw_i, int dtb_i, int alog_i, int dpar_i, int l,
                         float* Cout, const float* gptr, int epi,
                         const float* p2, const float* p3) {
        gemm(Ain, Win + (size_t)l*DX*DM, xz, BL, DX, DM, DM, DM, DX,
             nullptr, nullptr, 0, nullptr, nullptr, nullptr, 0, 0);
        for (int dir = 0; dir < 2; ++dir) {
            const float* convw = F(convw_i + 7*dir) + (size_t)l*DI*4;
            const float* convb = F(convb_i + 7*dir) + (size_t)l*DI;
            sdm_conv_silu<<<(BL*DI + 255)/256, 256, 0, stream>>>(xz, convw, convb, xc, xc16, dir);
            gemm(xc16, Wx + ((size_t)(xslot*2 + dir)*DEPTH + l)*DRN*DI, xdbl,
                 BL, DRN, DI, DI, DI, DRN, nullptr, nullptr, 0, nullptr, nullptr, nullptr, 0, 0);
            const float* dtw  = F(dtw_i + 7*dir)  + (size_t)l*DI*12;
            const float* dtbp = F(dtb_i + 7*dir)  + (size_t)l*DI;
            sdm_dtproj<<<(BL*DI + 255)/256, 256, 0, stream>>>(xdbl, dtw, dtbp, dtf);
            const float* alog = F(alog_i + 7*dir) + (size_t)l*DI*16;
            const float* dpar = F(dpar_i + 7*dir) + (size_t)l*DI;
            sdm_scan<<<(BSZ*DI*16 + 255)/256, 256, 0, stream>>>(
                xc, dtf, xdbl, alog, dpar, xz, ytot, dir, dir);
        }
        sdm_f32_to_f16<<<(BL*DI + 255)/256, 256, 0, stream>>>(ytot, y16, BL*DI);
        gemm(y16, Wout + (size_t)l*DM*DI, Cout, BL, DM, DI, DI, DI, DM,
             nullptr, gptr, 576, p2, p3, nullptr, 0, epi);
    };

    // ---- layer loop ----
    for (int l = 0; l < DEPTH; ++l) {
        // sem branch
        sdm_modulate_rms<<<256, 256, 0, stream>>>(
            h_sem, F(IN_NORM_SEM_W) + l*DM, msem + (size_t)l*BSZ*576,
            u16, nullptr, nullptr, nullptr, 0);
        run_mamba(u16, Win_s, Wout_s,
                  IN_SEM_CONV_W_F, IN_SEM_CONV_B_F, 0,
                  IN_SEM_DT_W_F, IN_SEM_DT_B_F, IN_SEM_ALOG_F, IN_SEM_D_F, l,
                  h_sem, msem + (size_t)l*BSZ*576 + 384, 5, nullptr, nullptr);
        // mask = sigmoid(h_sem @ mask_w^T + b)  (uses updated h_sem)
        sdm_f32_to_f16<<<(BL*DM + 255)/256, 256, 0, stream>>>(h_sem, hs16, BL*DM);
        gemm(hs16, Wmask + (size_t)l*DM*DM, maskb, BL, DM, DM, DM, DM, DM,
             F(IN_MASK_B) + l*DM, nullptr, 0, nullptr, nullptr, nullptr, 0, 3);
        // tex branch
        sdm_modulate_rms<<<256, 256, 0, stream>>>(
            h_tex, F(IN_NORM_TEX_W) + l*DM, mtex + (size_t)l*BSZ*576,
            tm16, xt16, sxt + (size_t)l*BSZ*DM, F(IN_ALPHA_X), l);
        gemm(tm16, Wtexz + (size_t)l*DM*DM, ztb, BL, DM, DM, DM, DM, DM,
             F(IN_TEXZ_B) + l*DM, szt + (size_t)l*BSZ*DM, DM, nullptr, nullptr,
             F(IN_ALPHA_Z), l, 4);
        run_mamba(xt16, Win_t, Wout_t,
                  IN_TEX_CONV_W_F, IN_TEX_CONV_B_F, 1,
                  IN_TEX_DT_W_F, IN_TEX_DT_B_F, IN_TEX_ALOG_F, IN_TEX_D_F, l,
                  h_tex, mtex + (size_t)l*BSZ*576 + 384, 6, ztb, maskb);
    }

    // ---- final layers + unpatchify + iDWT ----
    sdm_final_ln<<<256, 256, 0, stream>>>(h_sem, fsm, fin16);
    gemm(fin16, Wfs, osem, BL, DM, DM, DM, DM, DM,
         F(IN_FS_LIN_B), nullptr, 0, nullptr, nullptr, nullptr, 0, 1);
    sdm_final_ln<<<256, 256, 0, stream>>>(h_tex, ftm, fin16);
    gemm(fin16, Wft, otex, BL, 576, DM, DM, DM, 576,
         F(IN_FT_LIN_B), nullptr, 0, nullptr, nullptr, nullptr, 0, 1);
    sdm_unpatch_idwt<<<(BSZ*3*256*256 + 255)/256, 256, 0, stream>>>(osem, otex, (float*)d_out);
}
